// NeuralMemory_10144712753155
// MI455X (gfx1250) — matrix operand order repaired and checked by reading, not compiled, not run on hardware
//
#include <hip/hip_runtime.h>

// ---------------------------------------------------------------------------
// NeuralMemory (Titans-style) for MI455X / gfx1250.
//
// Math note: reference computes per-chunk grads with the SAME lmm_w and then
// means over chunks => g = (1/C) * full-batch gradient over all B*T tokens.
// Pipeline:
//   xm = [meta; x]                                   [B,T,D], T=2112
//   q/k/v = xm @ W^T, alr = sigmoid(xm @ wlr^T)*1e-2
//   fwd:  z1 = k@W0^T; h1 = k+silu(z1); z2 = h1@W1^T; h2 = h1+silu(z2)
//   bwd:  dh2 = (2*alr/D)*(h2-v); d2 = dh2*silu'(z2)
//         gW1 = d2^T@h1;  dh1 = d2@W1 + dh2; d1 = dh1*silu'(z1); gW0 = d1^T@k
//   AdamW(step1, zero state): Wn = W*(1-lr*wd) - lr*g/(|g|+eps), g = gsum/16
//   retrieval: r = lmm(Wn, q); SWA(window 512, 8 heads); slice rows t>=M.
//
// GEMMs: v_wmma_f32_16x16x32_bf16. Tiles staged to LDS as bf16 (f32->bf16
// converted once at staging time), fragments read as 16B ds_load_b128 runs,
// f32 accumulate. Element-wise / softmax / update math in f32.
// Scratch requirement: ~100 MB (11 [4224x512] f32 buffers + small).
// ---------------------------------------------------------------------------

typedef __attribute__((ext_vector_type(16))) __bf16 v16bf;
typedef __attribute__((ext_vector_type(8)))  __bf16 v8bf;
typedef __attribute__((ext_vector_type(8)))  float  v8f;

namespace nm {
constexpr int Bb  = 2;
constexpr int S   = 2048;
constexpr int D   = 512;
constexpr int Mm  = 64;          // meta tokens
constexpr int T   = Mm + S;      // 2112
constexpr int BT  = Bb * T;      // 4224  (multiple of 128)
constexpr int H   = 8;
constexpr int HD  = D / H;       // 64
constexpr int WIN = 512;
constexpr int C   = 16;
constexpr float MAX_ALR = 0.01f;
constexpr float LR = 1e-3f, WD = 1e-2f, EPS = 1e-8f;
constexpr int LDKH = 40;         // 32 bf16 + 8 pad -> 80B row stride
}

__device__ __forceinline__ float sigm_f(float x)  { return 1.0f / (1.0f + expf(-x)); }
__device__ __forceinline__ float silu_f(float x)  { return x * sigm_f(x); }
__device__ __forceinline__ float dsilu_f(float x) { float s = sigm_f(x); return s * (1.0f + x * (1.0f - s)); }

// ---- WMMA fragment builders from bf16 LDS tiles, CDNA5 16-bit layouts -----
// A 16x32 (MxK): lane l holds row m=l&15; elems 0..7 -> K = off+e,
// elems 8..15 -> K = 16+off+(e-8), off = (l>=16)?8:0.
__device__ __forceinline__ v16bf fragA_lds(const __bf16 (*As)[nm::LDKH], int mloc, int lane) {
  int m   = mloc + (lane & 15);
  int off = (lane >> 4) ? 8 : 0;
  v8bf lo = *(const v8bf*)&As[m][off];
  v8bf hi = *(const v8bf*)&As[m][16 + off];
  return __builtin_shufflevector(lo, hi, 0, 1, 2, 3, 4, 5, 6, 7,
                                 8, 9, 10, 11, 12, 13, 14, 15);
}
// B 32x16 (KxN): lane l holds col n=l&15; elems 0..7 -> K = off+e,
// elems 8..15 -> K = 16+off+(e-8), off = (l>=16)?8:0.
__device__ __forceinline__ v16bf fragB_lds(const __bf16 (*Bs)[nm::LDKH], int nloc, int lane) {
  int n   = nloc + (lane & 15);
  int off = (lane >> 4) ? 8 : 0;
  v8bf lo = *(const v8bf*)&Bs[n][off];
  v8bf hi = *(const v8bf*)&Bs[n][16 + off];
  return __builtin_shufflevector(lo, hi, 0, 1, 2, 3, 4, 5, 6, 7,
                                 8, 9, 10, 11, 12, 13, 14, 15);
}
// C/D 16x16 f32: lane l col n=l&15; VGPR r row m = r + ((l>=16)?8:0).
// Branch-free epilogue: C = alpha*acc + beta*Cadd (Cadd always valid).
__device__ __forceinline__ void store_tile(float* __restrict__ Cp, const float* __restrict__ Cadd,
                                           long ldc, long row0, long col0, v8f acc,
                                           int lane, float alpha, float beta) {
  int n = lane & 15;
  long r0 = row0 + ((lane >> 4) ? 8 : 0);
#pragma unroll
  for (int r = 0; r < 8; ++r) {
    long idx = (r0 + r) * ldc + col0 + n;
    Cp[idx] = alpha * acc[r] + beta * Cadd[idx];
  }
}

// Generic strided GEMM: C[m,n] = alpha * sum_k A(m,k)*B(n,k) + beta*Cadd[m,n].
// A(m,k)=A[m*sAm+k*sAk], B(n,k)=B[n*sBn+k*sBk].
// AKC: sAk==1 (else sAm==1); BKC: sBk==1 (else sBn==1).
// Md % 128 == 0, Nd % 64 == 0, Kd % 32 == 0 (true for every call here).
// 256 threads = 8 waves (4Mx2N); wave computes a 32x32 tile (4 wmma / K-step).
template <bool AKC, bool BKC>
__global__ __launch_bounds__(256) void k_gemm_wmma(
    const float* __restrict__ A, const float* __restrict__ B,
    const float* __restrict__ Cadd, float* __restrict__ Cout,
    int Md, int Nd, int Kd, long sAm, long sAk, long sBn, long sBk,
    float alpha, float beta) {
  using nm::LDKH;
  __shared__ __bf16 As[128][LDKH];   // 10 KB
  __shared__ __bf16 Bs[64][LDKH];    //  5 KB
  int tid  = threadIdx.x;
  int lane = tid & 31;
  int wave = tid >> 5;
  int tilesN = Nd >> 6;
  long rowB = (long)(blockIdx.x / tilesN) * 128;
  long colB = (long)(blockIdx.x % tilesN) * 64;
  int wm = (wave >> 1) * 32;
  int wn = (wave & 1) * 32;

  v8f acc00 = {}, acc01 = {}, acc10 = {}, acc11 = {};
  for (int kk = 0; kk < Kd; kk += 32) {
    __syncthreads();
    // ---- stage A tile: 128 rows x 32 k (16 f32 per thread, cvt to bf16) ----
    if (AKC) {
      int r = tid >> 1, kb = (tid & 1) * 16;
      const float* g = A + (rowB + r) * sAm + (kk + kb);
      float4 v0 = ((const float4*)g)[0], v1 = ((const float4*)g)[1],
             v2 = ((const float4*)g)[2], v3 = ((const float4*)g)[3];
      if (kk + 32 < Kd) __builtin_prefetch(g + 32, 0, 1);
      float buf[16];
      *(float4*)&buf[0] = v0; *(float4*)&buf[4]  = v1;
      *(float4*)&buf[8] = v2; *(float4*)&buf[12] = v3;
      __bf16 hb[16];
#pragma unroll
      for (int j = 0; j < 16; ++j) hb[j] = (__bf16)buf[j];
      *(v8bf*)&As[r][kb]     = *(v8bf*)&hb[0];
      *(v8bf*)&As[r][kb + 8] = *(v8bf*)&hb[8];
    } else {
      int k = tid >> 3, mb = (tid & 7) * 16;            // m contiguous (sAm==1)
      const float* g = A + (rowB + mb) + (long)(kk + k) * sAk;
      float4 v0 = ((const float4*)g)[0], v1 = ((const float4*)g)[1],
             v2 = ((const float4*)g)[2], v3 = ((const float4*)g)[3];
      if (kk + 32 < Kd) __builtin_prefetch(g + 32 * sAk, 0, 1);
      float buf[16];
      *(float4*)&buf[0] = v0; *(float4*)&buf[4]  = v1;
      *(float4*)&buf[8] = v2; *(float4*)&buf[12] = v3;
#pragma unroll
      for (int j = 0; j < 16; ++j) As[mb + j][k] = (__bf16)buf[j];
    }
    // ---- stage B tile: 64 rows x 32 k (8 f32 per thread, cvt to bf16) ----
    if (BKC) {
      int n = tid >> 2, kb = (tid & 3) * 8;
      const float* g = B + (colB + n) * sBn + (kk + kb);
      float4 v0 = ((const float4*)g)[0], v1 = ((const float4*)g)[1];
      if (kk + 32 < Kd) __builtin_prefetch(g + 32, 0, 1);
      float buf[8];
      *(float4*)&buf[0] = v0; *(float4*)&buf[4] = v1;
      __bf16 hb[8];
#pragma unroll
      for (int j = 0; j < 8; ++j) hb[j] = (__bf16)buf[j];
      *(v8bf*)&Bs[n][kb] = *(v8bf*)&hb[0];
    } else {
      int k = tid >> 3, nb = (tid & 7) * 8;             // n contiguous (sBn==1)
      const float* g = B + (colB + nb) + (long)(kk + k) * sBk;
      float4 v0 = ((const float4*)g)[0], v1 = ((const float4*)g)[1];
      if (kk + 32 < Kd) __builtin_prefetch(g + 32 * sBk, 0, 1);
      float buf[8];
      *(float4*)&buf[0] = v0; *(float4*)&buf[4] = v1;
#pragma unroll
      for (int j = 0; j < 8; ++j) Bs[nb + j][k] = (__bf16)buf[j];
    }
    __syncthreads();
    // ---- fragments straight from bf16 LDS + 4x WMMA ----
    v16bf a0 = fragA_lds(As, wm,      lane);
    v16bf a1 = fragA_lds(As, wm + 16, lane);
    v16bf b0 = fragB_lds(Bs, wn,      lane);
    v16bf b1 = fragB_lds(Bs, wn + 16, lane);
    acc00 = __builtin_amdgcn_wmma_f32_16x16x32_bf16(false, a0, false, b0, (short)0, acc00, false, false);
    acc01 = __builtin_amdgcn_wmma_f32_16x16x32_bf16(false, a0, false, b1, (short)0, acc01, false, false);
    acc10 = __builtin_amdgcn_wmma_f32_16x16x32_bf16(false, a1, false, b0, (short)0, acc10, false, false);
    acc11 = __builtin_amdgcn_wmma_f32_16x16x32_bf16(false, a1, false, b1, (short)0, acc11, false, false);
  }
  long row0 = rowB + wm, col0 = colB + wn;
  store_tile(Cout, Cadd, Nd, row0,      col0,      acc00, lane, alpha, beta);
  store_tile(Cout, Cadd, Nd, row0,      col0 + 16, acc01, lane, alpha, beta);
  store_tile(Cout, Cadd, Nd, row0 + 16, col0,      acc10, lane, alpha, beta);
  store_tile(Cout, Cadd, Nd, row0 + 16, col0 + 16, acc11, lane, alpha, beta);
}

// ---- element-wise kernels -------------------------------------------------
__global__ __launch_bounds__(256) void k_build_xm(const float* __restrict__ x,
                                                  const float* __restrict__ meta,
                                                  float* __restrict__ xm, int total) {
  int i = blockIdx.x * 256 + threadIdx.x;
  if (i >= total) return;
  int d = i & (nm::D - 1);
  int row = i >> 9;                 // D = 512
  int b = row / nm::T, t = row % nm::T;
  xm[i] = (t < nm::Mm) ? meta[t * nm::D + d]
                       : x[((long)(b * nm::S + (t - nm::Mm))) * nm::D + d];
}

__global__ __launch_bounds__(256) void k_alr(const float* __restrict__ xm,
                                             const float* __restrict__ wlr,
                                             float* __restrict__ alr) {
  int lane = threadIdx.x & 31;
  int tok = blockIdx.x * 8 + (threadIdx.x >> 5);
  const float* row = xm + (long)tok * nm::D;
  float s = 0.f;
  for (int i = lane; i < nm::D; i += 32) s += row[i] * wlr[i];
#pragma unroll
  for (int off = 16; off; off >>= 1) s += __shfl_xor(s, off);
  if (lane == 0) alr[tok] = sigm_f(s) * nm::MAX_ALR;
}

__global__ __launch_bounds__(256) void k_silu_res(const float* __restrict__ h0,
                                                  const float* __restrict__ z,
                                                  float* __restrict__ h, int total) {
  int i = blockIdx.x * 256 + threadIdx.x;
  if (i < total) h[i] = h0[i] + silu_f(z[i]);
}

__global__ __launch_bounds__(256) void k_dh2(const float* __restrict__ h2,
                                             const float* __restrict__ v,
                                             const float* __restrict__ z2,
                                             const float* __restrict__ alr,
                                             float* __restrict__ dh2,
                                             float* __restrict__ d2, int total) {
  int i = blockIdx.x * 256 + threadIdx.x;
  if (i >= total) return;
  float a = alr[i >> 9];
  float g = (2.0f / (float)nm::D) * a * (h2[i] - v[i]);
  dh2[i] = g;
  d2[i]  = g * dsilu_f(z2[i]);
}

__global__ __launch_bounds__(256) void k_dsilu_mul(float* __restrict__ dh1,
                                                   const float* __restrict__ z1, int total) {
  int i = blockIdx.x * 256 + threadIdx.x;
  if (i < total) dh1[i] *= dsilu_f(z1[i]);
}

__global__ __launch_bounds__(256) void k_adamw(const float* __restrict__ w,
                                               const float* __restrict__ gsum,
                                               float* __restrict__ wn, int total) {
  int i = blockIdx.x * 256 + threadIdx.x;
  if (i >= total) return;
  float g = gsum[i] * (1.0f / (float)nm::C);   // mean over chunks
  wn[i] = w[i] * (1.0f - nm::LR * nm::WD) - nm::LR * g / (fabsf(g) + nm::EPS);
}

// Sliding-window causal attention: one wave per (b, h, t); online softmax
// over keys j in [max(0, t-511), t]; each lane owns head-dims {lane, lane+32}.
__global__ __launch_bounds__(256) void k_attn(const float* __restrict__ qs,
                                              const float* __restrict__ ks,
                                              const float* __restrict__ vs,
                                              float* __restrict__ ao) {
  int lane = threadIdx.x & 31;
  int wg = blockIdx.x * 8 + (threadIdx.x >> 5);   // 0 .. BT*H-1
  int b = wg / (nm::T * nm::H);
  int rem = wg % (nm::T * nm::H);
  int t = rem / nm::H, h = rem % nm::H;
  long qoff = ((long)(b * nm::T + t)) * nm::D + h * nm::HD;
  float q0 = qs[qoff + lane], q1 = qs[qoff + lane + 32];
  float mval = -3.0e38f, lsum = 0.f, a0 = 0.f, a1 = 0.f;
  int jlo = t - (nm::WIN - 1); if (jlo < 0) jlo = 0;
  for (int j = jlo; j <= t; ++j) {
    long koff = ((long)(b * nm::T + j)) * nm::D + h * nm::HD;
    float s = q0 * ks[koff + lane] + q1 * ks[koff + lane + 32];
#pragma unroll
    for (int off = 16; off; off >>= 1) s += __shfl_xor(s, off);
    s *= 0.125f;                                  // 1/sqrt(64)
    float nmx = fmaxf(mval, s);
    float es = expf(mval - nmx);
    float p  = expf(s - nmx);
    lsum = lsum * es + p;
    a0 = a0 * es + p * vs[koff + lane];
    a1 = a1 * es + p * vs[koff + lane + 32];
    mval = nmx;
  }
  float inv = 1.0f / lsum;
  ao[qoff + lane]      = a0 * inv;
  ao[qoff + lane + 32] = a1 * inv;
}

__global__ __launch_bounds__(256) void k_slice(const float* __restrict__ full,
                                               float* __restrict__ out, int total) {
  int i = blockIdx.x * 256 + threadIdx.x;
  if (i >= total) return;
  int d = i & (nm::D - 1);
  int row = i >> 9;
  int b = row / nm::S, s = row % nm::S;
  out[i] = full[((long)(b * nm::T + nm::Mm + s)) * nm::D + d];
}

// ---------------------------------------------------------------------------
extern "C" void kernel_launch(void* const* d_in, const int* in_sizes, int n_in,
                              void* d_out, int out_size, void* d_ws, size_t ws_size,
                              hipStream_t stream) {
  using namespace nm;
  const float* x      = (const float*)d_in[0];
  const float* meta   = (const float*)d_in[1];
  const float* lmm_w  = (const float*)d_in[2];   // [2, D, D]
  const float* w_q    = (const float*)d_in[3];
  const float* w_k    = (const float*)d_in[4];
  const float* w_v    = (const float*)d_in[5];
  const float* w_lr   = (const float*)d_in[6];   // [1, D]
  const float* swa_wq = (const float*)d_in[7];
  const float* swa_wk = (const float*)d_in[8];
  const float* swa_wv = (const float*)d_in[9];
  const float* swa_wo = (const float*)d_in[10];

  const size_t TD = (size_t)BT * D;          // 2,162,688
  const size_t DD = (size_t)D * D;           // 262,144

  // scratch arena (f32), ~99.4 MB total; aliases noted where buffers are dead.
  float* ws  = (float*)d_ws;
  float* xm  = ws;            // [BT,D]
  float* q   = xm  + TD;
  float* kf  = q   + TD;
  float* vf  = kf  + TD;
  float* z1  = vf  + TD;
  float* z2  = z1  + TD;
  float* h1  = z2  + TD;
  float* h2  = h1  + TD;
  float* dh2 = h2  + TD;
  float* dh1 = dh2 + TD;      // becomes d1 in place
  float* d2  = dh1 + TD;
  float* alr = d2  + TD;      // [BT]
  float* gbuf= alr + BT;      // [2, D, D] raw grad sums (layer0, layer1)
  float* wn  = gbuf + 2 * DD; // [2, D, D] updated lmm weights
  // dead-buffer reuse:
  float* r1      = vf;   // v dead after k_dh2
  float* r2      = h2;   // h2 dead after k_dh2
  float* qs      = h1;   // h1 dead after gW1 GEMM
  float* ksb     = dh2;  // dh2 dead after dh1 GEMM
  float* vsb     = xm;   // xm dead after projections
  float* ao      = d2;   // d2 dead after gW1/dh1 GEMMs
  float* outfull = z1;   // z1 dead after d1 = dh1*silu'(z1)

  const dim3 blk(256);
  const int ew  = (int)((TD + 255) / 256);   // element-wise grid over [BT,D]
  // beta==0 calls pass Cadd=Cout (valid pointer; 0*finite == 0, 0xAA poison
  // pattern decodes to a finite float, and the scratch is ours anyway).
  auto gemm = [&](const float* A, long sAm, long sAk,
                  const float* Bp, long sBn, long sBk,
                  const float* Cadd, float* Cout, int Md, int Nd, int Kd,
                  float alpha, float beta) {
    dim3 grid((Md / 128) * (Nd / 64));
    const float* Ca = Cadd ? Cadd : Cout;
    bool akc = (sAk == 1), bkc = (sBk == 1);
    if (akc && bkc)
      k_gemm_wmma<true, true><<<grid, blk, 0, stream>>>(A, Bp, Ca, Cout, Md, Nd, Kd,
                                                        sAm, sAk, sBn, sBk, alpha, beta);
    else if (akc && !bkc)
      k_gemm_wmma<true, false><<<grid, blk, 0, stream>>>(A, Bp, Ca, Cout, Md, Nd, Kd,
                                                         sAm, sAk, sBn, sBk, alpha, beta);
    else if (!akc && bkc)
      k_gemm_wmma<false, true><<<grid, blk, 0, stream>>>(A, Bp, Ca, Cout, Md, Nd, Kd,
                                                         sAm, sAk, sBn, sBk, alpha, beta);
    else
      k_gemm_wmma<false, false><<<grid, blk, 0, stream>>>(A, Bp, Ca, Cout, Md, Nd, Kd,
                                                          sAm, sAk, sBn, sBk, alpha, beta);
  };

  // 1) xm = [meta; x]
  k_build_xm<<<ew, blk, 0, stream>>>(x, meta, xm, (int)TD);

  // 2) projections + adaptive lr
  gemm(xm, D, 1, w_q, D, 1, nullptr, q,  BT, D, D, 1.f, 0.f);
  gemm(xm, D, 1, w_k, D, 1, nullptr, kf, BT, D, D, 1.f, 0.f);
  gemm(xm, D, 1, w_v, D, 1, nullptr, vf, BT, D, D, 1.f, 0.f);
  k_alr<<<BT / 8, blk, 0, stream>>>(xm, w_lr, alr);

  // 3) LMM forward on keys
  gemm(kf, D, 1, lmm_w,      D, 1, nullptr, z1, BT, D, D, 1.f, 0.f);
  k_silu_res<<<ew, blk, 0, stream>>>(kf, z1, h1, (int)TD);
  gemm(h1, D, 1, lmm_w + DD, D, 1, nullptr, z2, BT, D, D, 1.f, 0.f);
  k_silu_res<<<ew, blk, 0, stream>>>(h1, z2, h2, (int)TD);

  // 4) backward: dh2/d2, gW1 = d2^T h1, dh1 = d2 @ W1 + dh2, d1, gW0 = d1^T k
  k_dh2<<<ew, blk, 0, stream>>>(h2, vf, z2, alr, dh2, d2, (int)TD);
  gemm(d2, 1, D, h1, 1, D, nullptr, gbuf + DD, D, D, BT, 1.f, 0.f);  // gW1[o,i]
  gemm(d2, D, 1, lmm_w + DD, 1, D, dh2, dh1, BT, D, D, 1.f, 1.f);    // dh1 = d2@W1 + dh2
  k_dsilu_mul<<<ew, blk, 0, stream>>>(dh1, z1, (int)TD);             // d1 in place
  gemm(dh1, 1, D, kf, 1, D, nullptr, gbuf, D, D, BT, 1.f, 0.f);      // gW0[o,i]

  // 5) AdamW step-1 from zero state
  k_adamw<<<(int)((2 * DD + 255) / 256), blk, 0, stream>>>(lmm_w, gbuf, wn, (int)(2 * DD));

  // 6) retrieval with updated weights
  gemm(q,  D, 1, wn,      D, 1, nullptr, z2, BT, D, D, 1.f, 0.f);
  k_silu_res<<<ew, blk, 0, stream>>>(q, z2, r1, (int)TD);
  gemm(r1, D, 1, wn + DD, D, 1, nullptr, z2, BT, D, D, 1.f, 0.f);
  k_silu_res<<<ew, blk, 0, stream>>>(r1, z2, r2, (int)TD);

  // 7) sliding-window attention
  gemm(r2, D, 1, swa_wq, D, 1, nullptr, qs,  BT, D, D, 1.f, 0.f);
  gemm(r2, D, 1, swa_wk, D, 1, nullptr, ksb, BT, D, D, 1.f, 0.f);
  gemm(r2, D, 1, swa_wv, D, 1, nullptr, vsb, BT, D, D, 1.f, 0.f);
  k_attn<<<BT * H / 8, blk, 0, stream>>>(qs, ksb, vsb, ao);
  gemm(ao, D, 1, swa_wo, D, 1, nullptr, outfull, BT, D, D, 1.f, 0.f);

  // 8) drop meta rows
  k_slice<<<(Bb * S * D) / 256, blk, 0, stream>>>(outfull, (float*)d_out, Bb * S * D);
}
